// ConvAttentionLayer_Decouple_19885698580723
// MI455X (gfx1250) — matrix operand order repaired and checked by reading, not compiled, not run on hardware
//
#include <hip/hip_runtime.h>
#include <hip/hip_bf16.h>

typedef _Float16 half_t;
typedef __attribute__((ext_vector_type(8)))  _Float16 v8h;
typedef __attribute__((ext_vector_type(16))) _Float16 v16h;
typedef __attribute__((ext_vector_type(8)))  float    v8f;

#define HW 4096
#define CC 256
#define NHEAD 8
#define ATTN_PER_HEAD ((size_t)HW * (size_t)HW)   // 16777216
#define TV_ELEMS ((size_t)HW * (size_t)CC)        // 1048576
#define ESTRIDE 4100                              // padded LDS row stride (floats)
#define FUSED_LDS_FLOATS (16 * ESTRIDE + 256 + 16 + 512)
#define FUSED_LDS_BYTES (FUSED_LDS_FLOATS * 4)    // 265536 < 320KB

// ---------------------------------------------------------------------------
// 3x3 conv, 256->64 ch, pad 1, fused (in + pos) add and output scale.
// Output stored TRANSPOSED per head: out[head][hw][8] f16, so a WMMA lane can
// fetch all D=8 values of one pixel with a single 16-byte load.
// grid: 64*4096/256 = 1024 blocks x 256 threads. Threads in a block share oc.
// ---------------------------------------------------------------------------
__global__ __launch_bounds__(256) void conv3x3_kernel(
    const float* __restrict__ in, const float* __restrict__ pos,
    const float* __restrict__ w, const float* __restrict__ bias,
    half_t* __restrict__ out_t, float out_scale) {
  int idx = blockIdx.x * 256 + threadIdx.x;     // [0, 64*4096)
  int oc = idx >> 12;
  int p  = idx & 4095;
  int y  = p >> 6;
  int x  = p & 63;
  float acc = bias[oc];
  const float* wbase = w + (size_t)oc * 256 * 9;
  for (int ic = 0; ic < 256; ++ic) {
    const float* ib = in  + (size_t)ic * HW;
    const float* pb = pos + (size_t)ic * HW;
    const float* wr = wbase + ic * 9;
    #pragma unroll
    for (int dy = -1; dy <= 1; ++dy) {
      int yy = y + dy;
      if (yy < 0 || yy > 63) continue;
      #pragma unroll
      for (int dx = -1; dx <= 1; ++dx) {
        int xx = x + dx;
        if (xx < 0 || xx > 63) continue;
        int o = yy * 64 + xx;
        acc += wr[(dy + 1) * 3 + (dx + 1)] * (ib[o] + pb[o]);
      }
    }
  }
  int head = oc >> 3;
  int d    = oc & 7;
  out_t[(size_t)head * HW * 8 + (size_t)p * 8 + d] = (half_t)(acc * out_scale);
}

// ---------------------------------------------------------------------------
// 1x1 conv, 256->256 ch (value path), f16 out in [ch][hw] (K-contiguous for
// the AV B-operand). grid 4096 x 256.
// ---------------------------------------------------------------------------
__global__ __launch_bounds__(256) void conv1x1_v_kernel(
    const float* __restrict__ val, const float* __restrict__ w,
    const float* __restrict__ bias, half_t* __restrict__ out) {
  int idx = blockIdx.x * 256 + threadIdx.x;     // [0, 256*4096)
  int oc = idx >> 12;
  int p  = idx & 4095;
  float acc = bias[oc];
  const float* wr = w + (size_t)oc * 256;
  for (int ic = 0; ic < 256; ++ic) acc += wr[ic] * val[(size_t)ic * HW + p];
  out[idx] = (half_t)acc;
}

// ---------------------------------------------------------------------------
// Fused attention: energy (WMMA) -> LDS, softmax in LDS, single NT write of
// normalized attn to HBM, AV GEMM (WMMA) with A from LDS, r-tile reduced via
// ds_add_f32. One workgroup = one (head, 16-query-row) slab.
// grid: 8*256 = 2048 blocks x 256 threads (8 waves), 265536 B dynamic LDS.
// ---------------------------------------------------------------------------
__global__ __launch_bounds__(256) void attn_fused_kernel(
    const half_t* __restrict__ qh_t, const half_t* __restrict__ kh_t,
    const half_t* __restrict__ vh, float* __restrict__ attn_out,
    float* __restrict__ rbuf) {
  extern __shared__ float smem[];
  float* E       = smem;                     // [16][ESTRIDE] energy slab
  float* scratch = smem + 16 * ESTRIDE;      // [256] reduction scratch
  float* rowstat = scratch + 256;            // [16] row max, then 1/sum
  float* rtile   = rowstat + 16;             // [16][32] r output tile

  int h   = blockIdx.x >> 8;
  int qt  = blockIdx.x & 255;
  int q0  = qt * 16;
  int tid = threadIdx.x;
  int wave = tid >> 5;
  int lane = tid & 31;
  int m  = lane & 15;
  int hi = lane >> 4;

  const half_t* qhh = qh_t + (size_t)h * HW * 8;   // [hw][8]
  const half_t* khh = kh_t + (size_t)h * HW * 8;   // [hw][8]
  const half_t* vhh = vh + (size_t)h * 32 * HW;    // [d][hw]

  rtile[tid] = 0.0f;
  rtile[tid + 256] = 0.0f;

  // ---- step 1: energy tiles into LDS (K=8 zero-padded f16 WMMA) ----------
  v16h a;
  #pragma unroll
  for (int j = 0; j < 16; ++j) a[j] = (half_t)0.0f;
  if (lane < 16) {
    v8h t = *(const v8h*)(qhh + (size_t)(q0 + lane) * 8);  // one b128 load
    #pragma unroll
    for (int j = 0; j < 8; ++j) a[j] = t[j];
  }
  for (int kt = wave; kt < 256; kt += 8) {
    int k0 = kt * 16;
    v16h b;
    #pragma unroll
    for (int j = 0; j < 16; ++j) b[j] = (half_t)0.0f;
    if (lane < 16) {
      v8h t = *(const v8h*)(khh + (size_t)(k0 + lane) * 8);  // one b128 load
      #pragma unroll
      for (int j = 0; j < 8; ++j) b[j] = t[j];
    }
    v8f acc = {};
    acc = __builtin_amdgcn_wmma_f32_16x16x32_f16(
        false, a, false, b, (short)0, acc, false, false);
    #pragma unroll
    for (int r = 0; r < 8; ++r)
      E[(r + 8 * hi) * ESTRIDE + k0 + m] = acc[r];
  }
  __syncthreads();

  // ---- step 2: softmax over each of the 16 rows ---------------------------
  {
    int row = tid >> 4;
    int t   = tid & 15;
    float* er = E + row * ESTRIDE;
    float mx = -3.402823466e+38f;
    for (int i = t; i < HW; i += 16) mx = fmaxf(mx, er[i]);
    scratch[tid] = mx;
    __syncthreads();
    if (t == 0) {
      float mm = scratch[row * 16];
      #pragma unroll
      for (int j = 1; j < 16; ++j) mm = fmaxf(mm, scratch[row * 16 + j]);
      rowstat[row] = mm;
    }
    __syncthreads();
    mx = rowstat[row];
    float s = 0.0f;
    for (int i = t; i < HW; i += 16) {
      float e = __expf(er[i] - mx);
      er[i] = e;
      s += e;
    }
    scratch[tid] = s;
    __syncthreads();
    if (t == 0) {
      float ss = 0.0f;
      #pragma unroll
      for (int j = 0; j < 16; ++j) ss += scratch[row * 16 + j];
      rowstat[row] = 1.0f / ss;
    }
    __syncthreads();
  }

  // normalize LDS in place + single coalesced non-temporal write of attn
  {
    float* ghead = attn_out + (size_t)h * ATTN_PER_HEAD;
    for (int r = 0; r < 16; ++r) {
      float inv = rowstat[r];
      float* erow = E + r * ESTRIDE;
      float* grow = ghead + (size_t)(q0 + r) * HW;
      for (int i = tid; i < HW; i += 256) {
        float v = erow[i] * inv;
        erow[i] = v;
        __builtin_nontemporal_store(v, grow + i);
      }
    }
  }
  __syncthreads();

  // ---- step 3: AV GEMM, A from LDS, waves split K=4096 --------------------
  {
    v8f acc0 = {};
    v8f acc1 = {};
    const float* arow = E + m * ESTRIDE;
    int kbeg = wave * 512;
    int kend = kbeg + 512;
    for (int k0 = kbeg; k0 < kend; k0 += 32) {
      v16h aa;
      #pragma unroll
      for (int j = 0; j < 8; ++j) aa[j]     = (half_t)arow[k0 + hi * 16 + j];
      #pragma unroll
      for (int j = 0; j < 8; ++j) aa[8 + j] = (half_t)arow[k0 + hi * 16 + 8 + j];
      v16h b0 = *(const v16h*)(vhh + (size_t)m * HW        + k0 + hi * 16);
      v16h b1 = *(const v16h*)(vhh + (size_t)(16 + m) * HW + k0 + hi * 16);
      acc0 = __builtin_amdgcn_wmma_f32_16x16x32_f16(
          false, aa, false, b0, (short)0, acc0, false, false);
      acc1 = __builtin_amdgcn_wmma_f32_16x16x32_f16(
          false, aa, false, b1, (short)0, acc1, false, false);
    }
    #pragma unroll
    for (int r = 0; r < 8; ++r) {
      int M = r + 8 * hi;
      atomicAdd(&rtile[M * 32 + m],      acc0[r]);
      atomicAdd(&rtile[M * 32 + 16 + m], acc1[r]);
    }
  }
  __syncthreads();

  // write r tile: rbuf[(h*32 + d)*HW + q0 + M]
  for (int idx = tid; idx < 512; idx += 256) {
    int M = idx >> 5;
    int d = idx & 31;
    rbuf[(size_t)(h * 32 + d) * HW + q0 + M] = rtile[idx];
  }
}

// ---------------------------------------------------------------------------
// out-proj 1x1 conv + residual + layernorm over channels; one block per pixel.
// tv[hw][c] = LN_c( value[c][hw] + bo[c] + sum_{c'} wo[c][c'] * r[c'][hw] )
// ---------------------------------------------------------------------------
__global__ __launch_bounds__(256) void out_ln_kernel(
    const float* __restrict__ rbuf, const float* __restrict__ wo,
    const float* __restrict__ bo, const float* __restrict__ value,
    const float* __restrict__ g, const float* __restrict__ bb,
    float* __restrict__ tv) {
  __shared__ float rcol[256];
  __shared__ float red[256];
  int hw = blockIdx.x;
  int c  = threadIdx.x;
  rcol[c] = rbuf[(size_t)c * HW + hw];
  __syncthreads();
  float acc = bo[c];
  const float* wr = wo + (size_t)c * 256;
  for (int k = 0; k < 256; ++k) acc += wr[k] * rcol[k];
  float val = value[(size_t)c * HW + hw] + acc;

  red[c] = val;
  __syncthreads();
  for (int s = 128; s > 0; s >>= 1) {
    if (c < s) red[c] += red[c + s];
    __syncthreads();
  }
  float mu = red[0] * (1.0f / 256.0f);
  __syncthreads();
  float d = val - mu;
  red[c] = d * d;
  __syncthreads();
  for (int s = 128; s > 0; s >>= 1) {
    if (c < s) red[c] += red[c + s];
    __syncthreads();
  }
  float var = red[0] * (1.0f / 256.0f);
  tv[(size_t)hw * 256 + c] = d * rsqrtf(var + 1e-5f) * g[c] + bb[c];
}

// ---------------------------------------------------------------------------
extern "C" void kernel_launch(void* const* d_in, const int* in_sizes, int n_in,
                              void* d_out, int out_size, void* d_ws, size_t ws_size,
                              hipStream_t stream) {
  (void)in_sizes; (void)n_in; (void)out_size; (void)ws_size;
  const float* query = (const float*)d_in[0];
  const float* key   = (const float*)d_in[1];
  const float* value = (const float*)d_in[2];
  const float* qpe   = (const float*)d_in[3];
  const float* kpe   = (const float*)d_in[4];
  const float* wq    = (const float*)d_in[5];
  const float* bq    = (const float*)d_in[6];
  const float* wk    = (const float*)d_in[7];
  const float* bk    = (const float*)d_in[8];
  const float* wv    = (const float*)d_in[9];
  const float* bv    = (const float*)d_in[10];
  const float* wo    = (const float*)d_in[11];
  const float* bo    = (const float*)d_in[12];
  const float* lng   = (const float*)d_in[13];
  const float* lnb   = (const float*)d_in[14];

  // workspace layout
  char* ws = (char*)d_ws;
  half_t* qh_t = (half_t*)(ws);                 // [8][4096][8] f16 = 512 KB
  half_t* kh_t = (half_t*)(ws + 524288);        // [8][4096][8] f16 = 512 KB
  half_t* vh   = (half_t*)(ws + 1048576);       // [256][4096] f16  = 2 MB
  float*  rbuf = (float*) (ws + 3145728);       // [256][4096] f32  = 4 MB

  float* tv   = (float*)d_out;                  // [4096][256]
  float* attn = (float*)d_out + TV_ELEMS;       // [8][4096][4096]

  const float scale = 0.35355339059327373f;     // D^-0.5, D=8

  static_assert(FUSED_LDS_BYTES <= 320 * 1024, "LDS slab exceeds WGP LDS");
  (void)hipFuncSetAttribute((const void*)attn_fused_kernel,
                            hipFuncAttributeMaxDynamicSharedMemorySize,
                            FUSED_LDS_BYTES);

  conv3x3_kernel<<<1024, 256, 0, stream>>>(query, qpe, wq, bq, qh_t, scale);
  conv3x3_kernel<<<1024, 256, 0, stream>>>(key,   kpe, wk, bk, kh_t, 1.0f);
  conv1x1_v_kernel<<<4096, 256, 0, stream>>>(value, wv, bv, vh);
  attn_fused_kernel<<<2048, 256, FUSED_LDS_BYTES, stream>>>(qh_t, kh_t, vh, attn, rbuf);
  out_ln_kernel<<<4096, 256, 0, stream>>>(rbuf, wo, bo, value, lng, lnb, tv);
}
